// BiLSTMCRF_43576738185728
// MI455X (gfx1250) — compile-verified, matrix operand order checked by reading, not run on hardware
//
#include <hip/hip_runtime.h>

#define T_LEN 512
#define BATCH 128
#define EDIM 256
#define HDIM 256
#define NTAG 32

typedef __attribute__((ext_vector_type(16))) __bf16 v16bf;
typedef __attribute__((ext_vector_type(8)))  __bf16 v8bf;
typedef __attribute__((ext_vector_type(8)))  float  v8f;
typedef __attribute__((ext_vector_type(4)))  unsigned int u32x4;
typedef __attribute__((ext_vector_type(8)))  int i32x8;
typedef __attribute__((ext_vector_type(4)))  int i32x4;

// ---------------------------------------------------------------------------
// WMMA helper (CDNA5: v_wmma_f32_16x16x32_bf16, D = A(16x32) * B(32x16) + C)
// ---------------------------------------------------------------------------
__device__ __forceinline__ v8f wmma_bf16(v16bf a, v16bf b, v8f c) {
    return __builtin_amdgcn_wmma_f32_16x16x32_bf16(
        false, a, false, b, (short)0, c, false, false);
}

// Load a 16x32 bf16 fragment per ISA 7.12.2 (16-bit A 16x32):
// lane row = lane&15, koff = (lane>>4)*8; two 16-byte chunks at k0+koff and
// k0+16+koff. Same layout serves B = W^T fragments (rows of W, N in place of M).
__device__ __forceinline__ v16bf load_frag(const __bf16* __restrict__ base,
                                           int ld, int row0, int k0) {
    int lane = threadIdx.x & 31;
    int r    = row0 + (lane & 15);
    int koff = (lane >> 4) << 3;
    const __bf16* p = base + (long)r * ld + k0 + koff;
    v8bf lo = *(const v8bf*)(p);
    v8bf hi = *(const v8bf*)(p + 16);
    return __builtin_shufflevector(lo, hi, 0,1,2,3,4,5,6,7,8,9,10,11,12,13,14,15);
}

__device__ __forceinline__ float sigmoidf(float x) {
    return 1.0f / (1.0f + expf(-x));
}

// ---------------------------------------------------------------------------
// Tensor Data Mover: DMA a 16-row x 256-elem bf16 tile (row stride 256) from
// global into LDS. D# per ISA 8.3/8.4: 2D tile, data_size=2B, groups 2/3 zero.
// Issued by one wave; completion via s_wait_tensorcnt.
// (6-arg builtin form: g0, g1, g2, g3, g4, cpol)
// ---------------------------------------------------------------------------
__device__ __forceinline__ void tdm_load_x_tile(const __bf16* gsrc,
                                                unsigned lds_byte_off) {
    unsigned long long ga = (unsigned long long)(const void*)gsrc;
    u32x4 g0;
    g0[0] = 1u;                                      // count=1 (valid D#)
    g0[1] = lds_byte_off;                            // lds_addr
    g0[2] = (unsigned)(ga & 0xffffffffu);            // global_addr[31:0]
    g0[3] = (unsigned)((ga >> 32) & 0x01ffffffu)     // global_addr[56:32]
            | (2u << 30);                            // type = 2 ("image")
    i32x8 g1;
    g1[0] = 0x00010000;          // data_size=1 -> 2 bytes; no multicast/pad
    g1[1] = (256 << 16);         // tensor_dim0 = 256 (bits 79:64 -> 0)
    g1[2] = 0;                   // tensor_dim0 hi; tensor_dim1 low16(65536)=0
    g1[3] = 1 | (256 << 16);     // tensor_dim1 hi16 = 1; tile_dim0 = 256
    g1[4] = 16;                  // tile_dim1 = 16; tile_dim2 = 0
    g1[5] = 256;                 // tensor_dim0_stride = 256
    g1[6] = 0;
    g1[7] = 0;
    i32x4 z4 = {0, 0, 0, 0};
    i32x8 z8 = {0, 0, 0, 0, 0, 0, 0, 0};
    __builtin_amdgcn_tensor_load_to_lds(g0, g1, z4, z4, z8, 0);
}

// ---------------------------------------------------------------------------
// 1) Embedding gather + bf16 convert
// ---------------------------------------------------------------------------
__global__ __launch_bounds__(256) void gather_kernel(
    const int* __restrict__ tokens, const float* __restrict__ embed,
    __bf16* __restrict__ xout) {
    long tb  = blockIdx.x;              // t*B + b, grid = T*B
    long tok = tokens[tb];
    xout[tb * EDIM + threadIdx.x] = (__bf16)embed[tok * EDIM + threadIdx.x];
}

__global__ __launch_bounds__(256) void cvt_kernel(
    const float* __restrict__ src, __bf16* __restrict__ dst, int n) {
    int i = blockIdx.x * blockDim.x + threadIdx.x;
    if (i < n) dst[i] = (__bf16)src[i];
}

// ---------------------------------------------------------------------------
// 2) Fused BiLSTM. grid.x = 16: bit3 = direction, bits0..2 = batch tile.
// 512 threads = 16 waves; wave w owns unit tile j0 = w*16 and computes the
// i/f/g/o gate tiles for its (16 batch x 16 unit) patch; c state in VGPRs.
// x_t tiles double-buffered in LDS via TDM; h_{t-1} in LDS (bf16).
// ---------------------------------------------------------------------------
__global__ __launch_bounds__(512) void bilstm_kernel(
    const __bf16* __restrict__ x,
    const __bf16* __restrict__ wih_f, const __bf16* __restrict__ whh_f,
    const float*  __restrict__ b_f,
    const __bf16* __restrict__ wih_b, const __bf16* __restrict__ whh_b,
    const float*  __restrict__ b_b,
    __bf16* __restrict__ h_f_out, __bf16* __restrict__ h_b_out) {
    // [ xbuf0 | xbuf1 | hbuf ] : 8KB + 8KB + 8KB
    __shared__ __align__(16) __bf16 smem[3 * 16 * EDIM];
    __bf16* hbuf = smem + 2 * 16 * EDIM;

    const int dir  = blockIdx.x >> 3;
    const int bt   = blockIdx.x & 7;        // batch rows [bt*16, bt*16+16)
    const int wave = threadIdx.x >> 5;      // 0..15
    const int lane = threadIdx.x & 31;
    const int j0   = wave * 16;             // unit tile within H
    const int n    = lane & 15;             // tile column (fixed per lane)

    const __bf16* wih  = dir ? wih_b : wih_f;
    const __bf16* whh  = dir ? whh_b : whh_f;
    const float*  bias = dir ? b_b   : b_f;
    __bf16*       hout = dir ? h_b_out : h_f_out;

    const float bi = bias[0 * HDIM + j0 + n];
    const float bf = bias[1 * HDIM + j0 + n];
    const float bg = bias[2 * HDIM + j0 + n];
    const float bo = bias[3 * HDIM + j0 + n];

    for (int i = threadIdx.x; i < 16 * HDIM; i += blockDim.x)
        hbuf[i] = (__bf16)0.0f;

    // prologue: DMA first x tile into xbuf0
    const int t0 = dir ? (T_LEN - 1) : 0;
    if (wave == 0) {
        tdm_load_x_tile(x + ((long)t0 * BATCH + bt * 16) * EDIM, 0u);
        __builtin_amdgcn_s_wait_tensorcnt(0);
    }
    __syncthreads();

    float c[8];
    #pragma unroll
    for (int e = 0; e < 8; ++e) c[e] = 0.0f;

    int cur = 0;
    for (int s = 0; s < T_LEN; ++s) {
        const int t   = dir ? (T_LEN - 1 - s) : s;
        const int nxt = cur ^ 1;

        // kick off DMA of the next timestep's x tile into the other buffer
        if (wave == 0 && s + 1 < T_LEN) {
            const int tn = dir ? (t - 1) : (t + 1);
            tdm_load_x_tile(x + ((long)tn * BATCH + bt * 16) * EDIM,
                            (unsigned)(nxt * 16 * EDIM * 2));
        }

        const __bf16* xt = smem + cur * 16 * EDIM;
        v8f acc_i = {}, acc_f = {}, acc_g = {}, acc_o = {};

        // K = 512 fused: first 256 from x_t (LDS), next 256 from h_{t-1} (LDS).
        // Weight fragments streamed from global (L2-resident) each iteration;
        // unroll 1 keeps register pressure low (no scratch spills).
        #pragma unroll 1
        for (int kt = 0; kt < 8; ++kt) {
            const int k0 = kt * 32;
            v16bf a = load_frag(xt, EDIM, 0, k0);
            v16bf w0 = load_frag(wih, EDIM, 0 * HDIM + j0, k0);
            v16bf w1 = load_frag(wih, EDIM, 1 * HDIM + j0, k0);
            v16bf w2 = load_frag(wih, EDIM, 2 * HDIM + j0, k0);
            v16bf w3 = load_frag(wih, EDIM, 3 * HDIM + j0, k0);
            acc_i = wmma_bf16(a, w0, acc_i);
            acc_f = wmma_bf16(a, w1, acc_f);
            acc_g = wmma_bf16(a, w2, acc_g);
            acc_o = wmma_bf16(a, w3, acc_o);
        }
        #pragma unroll 1
        for (int kt = 0; kt < 8; ++kt) {
            const int k0 = kt * 32;
            v16bf a = load_frag(hbuf, HDIM, 0, k0);
            v16bf w0 = load_frag(whh, HDIM, 0 * HDIM + j0, k0);
            v16bf w1 = load_frag(whh, HDIM, 1 * HDIM + j0, k0);
            v16bf w2 = load_frag(whh, HDIM, 2 * HDIM + j0, k0);
            v16bf w3 = load_frag(whh, HDIM, 3 * HDIM + j0, k0);
            acc_i = wmma_bf16(a, w0, acc_i);
            acc_f = wmma_bf16(a, w1, acc_f);
            acc_g = wmma_bf16(a, w2, acc_g);
            acc_o = wmma_bf16(a, w3, acc_o);
        }
        __syncthreads();   // all waves done reading hbuf and xbuf[cur]

        #pragma unroll
        for (int e = 0; e < 8; ++e) {
            const float gi = sigmoidf(acc_i[e] + bi);
            const float gf = sigmoidf(acc_f[e] + bf);
            const float gg = tanhf   (acc_g[e] + bg);
            const float go = sigmoidf(acc_o[e] + bo);
            c[e] = gf * c[e] + gi * gg;
            const float h = go * tanhf(c[e]);
            const int brow = e + ((lane >> 4) << 3);  // C/D layout row
            const __bf16 hbv = (__bf16)h;
            hbuf[brow * HDIM + j0 + n] = hbv;
            hout[((long)t * BATCH + bt * 16 + brow) * HDIM + j0 + n] = hbv;
        }

        // drain TDM for the next tile, then make h + x_{t+1} visible to all
        if (wave == 0 && s + 1 < T_LEN)
            __builtin_amdgcn_s_wait_tensorcnt(0);
        __syncthreads();
        cur = nxt;
    }
}

// ---------------------------------------------------------------------------
// 3) Projection: emissions(T*B,32) = [h_f | h_b](T*B,512) @ proj_w^T + b
// ---------------------------------------------------------------------------
__global__ __launch_bounds__(256) void proj_kernel(
    const __bf16* __restrict__ hf, const __bf16* __restrict__ hb,
    const __bf16* __restrict__ pw, const float* __restrict__ pb,
    float* __restrict__ em) {
    const int wave = threadIdx.x >> 5;
    const int lane = threadIdx.x & 31;
    const int row0 = (blockIdx.x * 8 + wave) * 16;   // grid = (T*B/16)/8 = 512

    v8f acc0 = {}, acc1 = {};
    #pragma unroll 2
    for (int kt = 0; kt < 16; ++kt) {
        const int k0 = kt * 32;
        v16bf a = (k0 < HDIM) ? load_frag(hf, HDIM, row0, k0)
                              : load_frag(hb, HDIM, row0, k0 - HDIM);
        v16bf b0 = load_frag(pw, 2 * HDIM, 0,  k0);
        v16bf b1 = load_frag(pw, 2 * HDIM, 16, k0);
        acc0 = wmma_bf16(a, b0, acc0);
        acc1 = wmma_bf16(a, b1, acc1);
    }
    const int nn  = lane & 15;
    const float pb0 = pb[nn], pb1 = pb[16 + nn];
    #pragma unroll
    for (int e = 0; e < 8; ++e) {
        const long r = row0 + e + ((lane >> 4) << 3);
        em[r * NTAG + nn]      = acc0[e] + pb0;
        em[r * NTAG + 16 + nn] = acc1[e] + pb1;
    }
}

// ---------------------------------------------------------------------------
// 4) CRF NLL. One wave32 per batch element; lane == tag (NTAG == 32).
// ---------------------------------------------------------------------------
__global__ void zero_kernel(float* __restrict__ out) {
    if (threadIdx.x == 0 && blockIdx.x == 0) out[0] = 0.0f;
}

__global__ __launch_bounds__(256) void crf_kernel(
    const float* __restrict__ em, const int* __restrict__ tags,
    const bool* __restrict__ mask,
    const float* __restrict__ start_trans, const float* __restrict__ end_trans,
    const float* __restrict__ trans, float* __restrict__ out) {
    __shared__ float tr[NTAG * NTAG];
    for (int i = threadIdx.x; i < NTAG * NTAG; i += blockDim.x) tr[i] = trans[i];
    __syncthreads();

    const int wave = threadIdx.x >> 5;
    const int lane = threadIdx.x & 31;
    const int b    = blockIdx.x * 8 + wave;   // grid = 16 -> b in [0,128)

    // ---- partition function (forward algorithm); alpha[lane] per tag ----
    float alpha = start_trans[lane] + em[(long)b * NTAG + lane];
    for (int t = 1; t < T_LEN; ++t) {
        const float emv = em[((long)t * BATCH + b) * NTAG + lane];
        const bool  m   = mask[t * BATCH + b];
        float mx = -3.0e38f;
        #pragma unroll
        for (int i = 0; i < 32; ++i)
            mx = fmaxf(mx, __shfl(alpha, i, 32) + tr[i * NTAG + lane]);
        float sum = 0.0f;
        #pragma unroll
        for (int i = 0; i < 32; ++i)
            sum += expf(__shfl(alpha, i, 32) + tr[i * NTAG + lane] - mx);
        const float nxt = mx + logf(sum) + emv;
        alpha = m ? nxt : alpha;
    }
    float v  = alpha + end_trans[lane];
    float mx = v;
    #pragma unroll
    for (int off = 16; off; off >>= 1) mx = fmaxf(mx, __shfl_xor(mx, off, 32));
    float s = expf(v - mx);
    #pragma unroll
    for (int off = 16; off; off >>= 1) s += __shfl_xor(s, off, 32);
    const float norm = mx + logf(s);

    // ---- gold path score, lane-parallel over time ----
    float part = 0.0f;
    int   cnt  = 0;
    for (int t = lane; t < T_LEN; t += 32) {
        const bool m = mask[t * BATCH + b];
        cnt += m ? 1 : 0;
        if (t >= 1 && m) {
            const int tp = tags[(t - 1) * BATCH + b];
            const int tc = tags[t * BATCH + b];
            part += tr[tp * NTAG + tc] + em[((long)t * BATCH + b) * NTAG + tc];
        }
    }
    #pragma unroll
    for (int off = 16; off; off >>= 1) {
        part += __shfl_xor(part, off, 32);
        cnt  += __shfl_xor(cnt,  off, 32);
    }
    if (lane == 0) {
        const int t0tag = tags[b];
        float score = start_trans[t0tag] + em[(long)b * NTAG + t0tag] + part;
        const int last = tags[(cnt - 1) * BATCH + b];
        score += end_trans[last];
        atomicAdd(out, norm - score);   // nll = sum_b (norm - score)
    }
}

// ---------------------------------------------------------------------------
// Launcher
// ---------------------------------------------------------------------------
extern "C" void kernel_launch(void* const* d_in, const int* in_sizes, int n_in,
                              void* d_out, int out_size, void* d_ws, size_t ws_size,
                              hipStream_t stream) {
    const int*   tokens      = (const int*)  d_in[0];
    const int*   tags        = (const int*)  d_in[1];
    const bool*  mask        = (const bool*) d_in[2];
    const float* embed       = (const float*)d_in[3];
    const float* w_ih_f      = (const float*)d_in[4];
    const float* w_hh_f      = (const float*)d_in[5];
    const float* b_f         = (const float*)d_in[6];
    const float* w_ih_b      = (const float*)d_in[7];
    const float* w_hh_b      = (const float*)d_in[8];
    const float* b_b         = (const float*)d_in[9];
    const float* proj_w      = (const float*)d_in[10];
    const float* proj_b      = (const float*)d_in[11];
    const float* start_trans = (const float*)d_in[12];
    const float* end_trans   = (const float*)d_in[13];
    const float* trans       = (const float*)d_in[14];
    float* out = (float*)d_out;

    char* w = (char*)d_ws;
    __bf16* x_bf  = (__bf16*)w; w += (size_t)T_LEN * BATCH * EDIM * 2;   // 33.5 MB
    __bf16* wihf  = (__bf16*)w; w += (size_t)4 * HDIM * EDIM * 2;        // 512 KB
    __bf16* whhf  = (__bf16*)w; w += (size_t)4 * HDIM * HDIM * 2;
    __bf16* wihb  = (__bf16*)w; w += (size_t)4 * HDIM * EDIM * 2;
    __bf16* whhb  = (__bf16*)w; w += (size_t)4 * HDIM * HDIM * 2;
    __bf16* pwb   = (__bf16*)w; w += (size_t)NTAG * 2 * HDIM * 2;        // 32 KB
    __bf16* hf    = (__bf16*)w; w += (size_t)T_LEN * BATCH * HDIM * 2;   // 33.5 MB
    __bf16* hb    = (__bf16*)w; w += (size_t)T_LEN * BATCH * HDIM * 2;
    float*  em    = (float*)w;  w += (size_t)T_LEN * BATCH * NTAG * 4;   // 8.4 MB

    gather_kernel<<<T_LEN * BATCH, 256, 0, stream>>>(tokens, embed, x_bf);

    const int nw = 4 * HDIM * EDIM;
    cvt_kernel<<<(nw + 255) / 256, 256, 0, stream>>>(w_ih_f, wihf, nw);
    cvt_kernel<<<(nw + 255) / 256, 256, 0, stream>>>(w_hh_f, whhf, nw);
    cvt_kernel<<<(nw + 255) / 256, 256, 0, stream>>>(w_ih_b, wihb, nw);
    cvt_kernel<<<(nw + 255) / 256, 256, 0, stream>>>(w_hh_b, whhb, nw);
    const int np = NTAG * 2 * HDIM;
    cvt_kernel<<<(np + 255) / 256, 256, 0, stream>>>(proj_w, pwb, np);

    bilstm_kernel<<<16, 512, 0, stream>>>(x_bf, wihf, whhf, b_f,
                                          wihb, whhb, b_b, hf, hb);

    proj_kernel<<<(T_LEN * BATCH / 16) / 8, 256, 0, stream>>>(hf, hb, pwb, proj_b, em);

    zero_kernel<<<1, 32, 0, stream>>>(out);
    crf_kernel<<<BATCH / 8, 256, 0, stream>>>(em, tags, mask, start_trans,
                                              end_trans, trans, out);
}
